// MultiLanguageOutputHead_52003464020428
// MI455X (gfx1250) — matrix operand order repaired and checked by reading, not compiled, not run on hardware
//
#include <hip/hip_runtime.h>
#include <hip/hip_bf16.h>

typedef __attribute__((ext_vector_type(16))) __bf16 v16bf;
typedef __attribute__((ext_vector_type(8)))  float  v8f;

#define B_   8
#define S_   128
#define H_   512
#define SHV  25000
#define LAV  2500
#define GENV 27500

// flat output offsets (floats), in reference return order
#define O_SHARED 0L
#define O_LANG   25600000L
#define O_GEN    28160000L
#define O_DS     56320000L
#define O_DR     81920000L
#define O_TOOL   107520000L
#define O_ROUTE  133120000L

static __device__ __forceinline__ v16bf cvt16(const float* f) {
    v16bf r;
#pragma unroll
    for (int i = 0; i < 16; ++i) r[i] = (__bf16)f[i];
    return r;
}

// A fragment: 16x32 bf16 tile of hidden[b] at (s0, k0).
// lane L: M = s0 + (L&15). K = k0 + 16*hi + i
static __device__ __forceinline__ v16bf load_a_frag(const float* __restrict__ Arow,
                                                    int k0, int hi) {
    float t[16];
    const float4* p0 = (const float4*)(Arow + k0 + 16 * hi);
    const float4* p1 = (const float4*)(Arow + k0 + 16 * hi + 8);
    float4 a0 = p0[0], a1 = p0[1], a2 = p1[0], a3 = p1[1];
    t[0]=a0.x; t[1]=a0.y; t[2]=a0.z; t[3]=a0.w;
    t[4]=a1.x; t[5]=a1.y; t[6]=a1.z; t[7]=a1.w;
    t[8]=a2.x; t[9]=a2.y; t[10]=a2.z; t[11]=a2.w;
    t[12]=a3.x; t[13]=a3.y; t[14]=a3.z; t[15]=a3.w;
    return cvt16(t);
}

// B fragment (unguarded): B[k,n] = W[v, k]; lane L: n = v0+(L&15),
// K = k0 + 16*hi + i -> 16 consecutive f32 from one weight row (4x b128 loads).
static __device__ __forceinline__ v16bf load_b_fast(const float* __restrict__ W,
                                                    int v, int k0, int hi) {
    float t[16];
    const float4* p = (const float4*)(W + (size_t)v * H_ + k0 + 16 * hi);
    float4 b0 = p[0], b1 = p[1], b2 = p[2], b3 = p[3];
    t[0]=b0.x; t[1]=b0.y; t[2]=b0.z; t[3]=b0.w;
    t[4]=b1.x; t[5]=b1.y; t[6]=b1.z; t[7]=b1.w;
    t[8]=b2.x; t[9]=b2.y; t[10]=b2.z; t[11]=b2.w;
    t[12]=b3.x; t[13]=b3.y; t[14]=b3.z; t[15]=b3.w;
    return cvt16(t);
}

static __device__ __forceinline__ v16bf load_b_guard(const float* __restrict__ W,
                                                     int v, int V, int k0, int hi) {
    if (v < V) return load_b_fast(W, v, k0, hi);
    v16bf r;
#pragma unroll
    for (int i = 0; i < 16; ++i) r[i] = (__bf16)0.0f;
    return r;
}

// store 16x16 f32 C/D tile (non-temporal): lane L -> N = n0+(L&15); VGPR r -> M = r+8*hi
template <bool GUARD>
static __device__ __forceinline__ void store_tile(float* __restrict__ out, long rowBase,
                                                  long rstride, int n0, int V,
                                                  const v8f acc, int lo, int hi) {
    int n = n0 + lo;
    if (GUARD && n >= V) return;
    float* p = out + rowBase * rstride + n + (long)(8 * hi) * rstride;
#pragma unroll
    for (int r = 0; r < 8; ++r) {
        __builtin_nontemporal_store(acc[r], p);
        p += rstride;
    }
}

template <bool GUARD>
static __device__ __forceinline__ void gemm_strip(const float* __restrict__ Arow,
                                                  const float* __restrict__ W,
                                                  int V, int n0w, int lo, int hi,
                                                  v8f acc[4]) {
    const int v0 = n0w + lo;
#pragma unroll 4
    for (int kk = 0; kk < 16; ++kk) {
        const int k0 = kk * 32;
        // Stage A and all four B fragments in distinct registers first, then
        // issue the 4 WMMAs back-to-back (no WAR hazard NOPs between them).
        v16bf a  = load_a_frag(Arow, k0, hi);
        v16bf b0 = GUARD ? load_b_guard(W, v0,      V, k0, hi) : load_b_fast(W, v0,      k0, hi);
        v16bf b1 = GUARD ? load_b_guard(W, v0 + 16, V, k0, hi) : load_b_fast(W, v0 + 16, k0, hi);
        v16bf b2 = GUARD ? load_b_guard(W, v0 + 32, V, k0, hi) : load_b_fast(W, v0 + 32, k0, hi);
        v16bf b3 = GUARD ? load_b_guard(W, v0 + 48, V, k0, hi) : load_b_fast(W, v0 + 48, k0, hi);
        acc[0] = __builtin_amdgcn_wmma_f32_16x16x32_bf16(false, a, false, b0, (short)0, acc[0], false, false);
        acc[1] = __builtin_amdgcn_wmma_f32_16x16x32_bf16(false, a, false, b1, (short)0, acc[1], false, false);
        acc[2] = __builtin_amdgcn_wmma_f32_16x16x32_bf16(false, a, false, b2, (short)0, acc[2], false, false);
        acc[3] = __builtin_amdgcn_wmma_f32_16x16x32_bf16(false, a, false, b3, (short)0, acc[3], false, false);
    }
}

__global__ __launch_bounds__(128)
void mloh_heads_kernel(const float* __restrict__ hidden,
                       const int* __restrict__ lang_ids,
                       const int* __restrict__ task_ids,
                       const float* __restrict__ shared_W,
                       const float* __restrict__ lang_W,
                       const float* __restrict__ dsW,
                       const float* __restrict__ drW,
                       const float* __restrict__ toolW,
                       float* __restrict__ out) {
    const int head = blockIdx.z;          // 0..4
    const int by   = blockIdx.y;          // b*8 + s_tile
    const int b    = by >> 3;
    const int s0   = (by & 7) << 4;
    const int wave = threadIdx.x >> 5;
    const int lane = threadIdx.x & 31;
    const int lo   = lane & 15;
    const int hi   = lane >> 4;

    int V;
    long rstride;
    const float* W;
    float* outp;
    float* gen = nullptr;                  // optional second (generation) store
    int genCol = 0;
    bool compute = true;
    int nblocks;

    switch (head) {
        case 0:
            V = SHV; W = shared_W; outp = out + O_SHARED; rstride = SHV;
            gen = out + O_GEN; genCol = 0; nblocks = 98; break;
        case 1:
            V = LAV; W = lang_W + (size_t)lang_ids[b] * LAV * H_;
            outp = out + O_LANG; rstride = LAV;
            gen = out + O_GEN; genCol = SHV; nblocks = 10; break;
        case 2:
            V = SHV; W = dsW; outp = out + O_DS; rstride = SHV;
            compute = (task_ids[b] == 1); nblocks = 98; break;
        case 3:
            V = SHV; W = drW; outp = out + O_DR; rstride = SHV;
            compute = (task_ids[b] == 1); nblocks = 98; break;
        default:
            V = SHV; W = toolW; outp = out + O_TOOL; rstride = SHV;
            compute = (task_ids[b] == 2); nblocks = 98; break;
    }
    if ((int)blockIdx.x >= nblocks) return;

    const int  n0w      = blockIdx.x * 256 + wave * 64;  // wave's 16x64 strip
    const long rowBase  = (long)b * S_ + s0;
    const bool interior = (n0w + 64 <= V);               // wave-uniform

    if (!compute) {
        v8f z = {};
#pragma unroll
        for (int j = 0; j < 4; ++j)
            store_tile<true>(outp, rowBase, rstride, n0w + 16 * j, V, z, lo, hi);
        return;
    }

    const float* Arow = hidden + ((size_t)(b * S_ + s0 + lo)) * H_;
    v8f acc[4] = {};

    if (interior) {
        gemm_strip<false>(Arow, W, V, n0w, lo, hi, acc);
#pragma unroll
        for (int j = 0; j < 4; ++j) {
            const int n0 = n0w + 16 * j;
            store_tile<false>(outp, rowBase, rstride, n0, V, acc[j], lo, hi);
            if (gen)
                store_tile<false>(gen, rowBase, GENV, genCol + n0, 0, acc[j], lo, hi);
        }
    } else {
        gemm_strip<true>(Arow, W, V, n0w, lo, hi, acc);
#pragma unroll
        for (int j = 0; j < 4; ++j) {
            const int n0 = n0w + 16 * j;
            store_tile<true>(outp, rowBase, rstride, n0, V, acc[j], lo, hi);
            if (gen)
                store_tile<true>(gen, rowBase, GENV, genCol + n0, genCol + V, acc[j], lo, hi);
        }
    }
}

// pooled[b,h] = mean_s hidden[b,s,h]
__global__ void mloh_pool_kernel(const float* __restrict__ hidden,
                                 float* __restrict__ pooled) {
    int t = blockIdx.x * blockDim.x + threadIdx.x;
    if (t >= B_ * H_) return;
    int b = t >> 9, h = t & (H_ - 1);
    const float* p = hidden + (size_t)b * S_ * H_ + h;
    float s = 0.f;
#pragma unroll 8
    for (int i = 0; i < S_; ++i) s += p[(size_t)i * H_];
    pooled[t] = s * (1.0f / S_);
}

// routing[b,t] = pooled[b] . router_W[t] + router_b[t]   (one wave per output)
__global__ void mloh_router_kernel(const float* __restrict__ pooled,
                                   const float* __restrict__ rW,
                                   const float* __restrict__ rb,
                                   float* __restrict__ outr) {
    int w = (blockIdx.x * blockDim.x + threadIdx.x) >> 5;
    int lane = threadIdx.x & 31;
    if (w >= B_ * 5) return;
    int b = w / 5, t = w % 5;
    float s = 0.f;
    for (int h = lane; h < H_; h += 32)
        s += pooled[b * H_ + h] * rW[t * H_ + h];
#pragma unroll
    for (int off = 16; off; off >>= 1) s += __shfl_xor(s, off, 32);
    if (lane == 0) outr[b * 5 + t] = s + rb[t];
}

extern "C" void kernel_launch(void* const* d_in, const int* in_sizes, int n_in,
                              void* d_out, int out_size, void* d_ws, size_t ws_size,
                              hipStream_t stream) {
    const float* hidden   = (const float*)d_in[0];
    const int*   lang_ids = (const int*)d_in[1];
    const int*   task_ids = (const int*)d_in[2];
    const float* shared_W = (const float*)d_in[3];
    const float* lang_W   = (const float*)d_in[4];
    const float* dsW      = (const float*)d_in[5];
    const float* drW      = (const float*)d_in[6];
    const float* toolW    = (const float*)d_in[7];
    const float* rW       = (const float*)d_in[8];
    const float* rb       = (const float*)d_in[9];
    float* out    = (float*)d_out;
    float* pooled = (float*)d_ws;   // 8*512 floats = 16 KB scratch

    dim3 grid(98, B_ * 8, 5);
    dim3 block(128);
    mloh_heads_kernel<<<grid, block, 0, stream>>>(hidden, lang_ids, task_ids,
                                                  shared_W, lang_W, dsW, drW, toolW, out);
    mloh_pool_kernel<<<(B_ * H_ + 255) / 256, 256, 0, stream>>>(hidden, pooled);
    mloh_router_kernel<<<5, 256, 0, stream>>>(pooled, rW, rb, out + O_ROUTE);
}
